// HNN_structure_embed_3607772529330
// MI455X (gfx1250) — compile-verified, matrix operand order checked by reading, not run on hardware
//
#include <hip/hip_runtime.h>
#include <hip/hip_bf16.h>
#include <math.h>

// ---------------------------------------------------------------------------
// HNN structured dynamics for MI455X (gfx1250, wave32, WMMA).
//
// Heavy work = six 32768x1024 @ 1024x1024 GEMMs -> v_wmma_f32_16x16x32_bf16.
// Each wave register-blocks a 32x64 output tile (2x4 fragments, 8 WMMAs per
// K-step) so operand traffic is ~0.047 B/FLOP instead of 0.125 B/FLOP.
// Weights are converted to bf16 and pre-packed into the per-lane WMMA
// B-fragment layout once per launch; activations are bf16 in d_ws.
// Workspace requirement: ~400 MB (6 x 64MB activations + 10MB packed weights
// + ~5MB fp32 side buffers).
// ---------------------------------------------------------------------------

typedef __bf16 bf16;
typedef __attribute__((ext_vector_type(16))) __bf16 v16bf;
typedef __attribute__((ext_vector_type(8)))  __bf16 v8bf;
typedef __attribute__((ext_vector_type(8)))  float  v8f;

#define BS    32768
#define HID   1024
#define XDIM  10      // 3*N+1, N=3

// ---------------------------------------------------------------------------
// Pack a 1024x1024 fp32 weight (optionally transposed) into bf16 WMMA
// B-fragments. Fragment (kk,nn) covers K=[kk*32,kk*32+32), Ncols=[nn*16,+16).
// Lane L, element j  ->  K = kk*32 + (L>=16?8:0) + (j<8 ? j : j+8),
//                        n = nn*16 + (L&15)         (mirror of the A layout).
// Flat layout: Bp[((kk*64+nn)*32 + L)*16 + j]
// ---------------------------------------------------------------------------
__global__ void k_pack(const float* __restrict__ W, bf16* __restrict__ Bp,
                       int transpose)
{
  int idx = blockIdx.x * blockDim.x + threadIdx.x;      // 0 .. 1024*1024-1
  int j   = idx & 15;
  int L   = (idx >> 4) & 31;
  int nn  = (idx >> 9) & 63;
  int kk  = idx >> 15;                                  // 0..31
  int K   = kk * 32 + ((L >> 4) << 3) + (j < 8 ? j : j + 8);
  int n   = nn * 16 + (L & 15);
  float v = transpose ? W[(size_t)n * HID + K] : W[(size_t)K * HID + n];
  Bp[idx] = (bf16)v;
}

// ---------------------------------------------------------------------------
// Layer-1 forward for all three nets: h1 = tanh(csq @ W1 + b1), K = 6.
// One thread per (row, j).
// ---------------------------------------------------------------------------
__global__ void k_layer1(const float* __restrict__ x,
                         const float* __restrict__ MW1, const float* __restrict__ Mb1,
                         const float* __restrict__ VW1, const float* __restrict__ Vb1,
                         const float* __restrict__ GW1, const float* __restrict__ Gb1,
                         bf16* __restrict__ h1m, bf16* __restrict__ h1v,
                         bf16* __restrict__ h1g)
{
  int idx = blockIdx.x * blockDim.x + threadIdx.x;
  int j   = idx & (HID - 1);
  int row = idx >> 10;
  float c[6];
#pragma unroll
  for (int i = 0; i < 6; ++i) c[i] = x[(size_t)row * XDIM + i];
  float zm = Mb1[j], zv = Vb1[j], zg = Gb1[j];
#pragma unroll
  for (int i = 0; i < 6; ++i) {
    zm += c[i] * MW1[i * HID + j];
    zv += c[i] * VW1[i * HID + j];
    zg += c[i] * GW1[i * HID + j];
  }
  h1m[idx] = (bf16)tanhf(zm);
  h1v[idx] = (bf16)tanhf(zv);
  h1g[idx] = (bf16)tanhf(zg);
}

// ---------------------------------------------------------------------------
// WMMA GEMM: C[BS,1024] = act(A[BS,1024] @ B[1024,1024] (+bias))
// MODE 0: plain; MODE 1: bias + tanh.
// Grid: (BS/128, 1024/128). 8 waves/WG arranged 4 (rowbands) x 2 (colbands);
// each wave register-blocks a 32x64 tile = 2x4 16x16 fragments.
// ---------------------------------------------------------------------------
template <int MODE>
__global__ __launch_bounds__(256) void k_gemm(const bf16* __restrict__ A,
                                              const bf16* __restrict__ Bp,
                                              const float* __restrict__ bias,
                                              bf16* __restrict__ C)
{
  const int lane    = threadIdx.x & 31;
  const int wave    = threadIdx.x >> 5;
  const int rowband = wave >> 1;                         // 0..3
  const int colband = wave & 1;                          // 0..1
  const int wrow0   = blockIdx.x * 128 + rowband * 32;   // wave's first row
  const int ncb0    = blockIdx.y * 8 + colband * 4;      // first 16-col block
  const int khalf   = (lane >> 4) * 8;

  const bf16* arow0 = A + (size_t)(wrow0 + (lane & 15)) * HID + khalf;
  const bf16* bbase = Bp + (size_t)ncb0 * 512 + (size_t)lane * 16;

  v8f acc[2][4];
  v8f zero = {};
#pragma unroll
  for (int rb = 0; rb < 2; ++rb)
#pragma unroll
    for (int cb = 0; cb < 4; ++cb) acc[rb][cb] = zero;

#pragma unroll 2
  for (int kk = 0; kk < HID / 32; ++kk) {
    // A fragments: two contiguous 16B runs per lane, per ISA 16-bit A layout
    v16bf a[2];
#pragma unroll
    for (int rb = 0; rb < 2; ++rb) {
      v8bf alo = *(const v8bf*)(arow0 + (size_t)rb * 16 * HID + kk * 32);
      v8bf ahi = *(const v8bf*)(arow0 + (size_t)rb * 16 * HID + kk * 32 + 16);
#pragma unroll
      for (int i = 0; i < 8; ++i) { a[rb][i] = alo[i]; a[rb][i + 8] = ahi[i]; }
    }
    // B fragments: pre-packed, one 32B load per lane each
    v16bf b[4];
#pragma unroll
    for (int cb = 0; cb < 4; ++cb)
      b[cb] = *(const v16bf*)(bbase + (size_t)(kk * 64 + cb) * 512);
    __builtin_prefetch(bbase + (size_t)((kk + 1) * 64) * 512, 0, 0);
#pragma unroll
    for (int rb = 0; rb < 2; ++rb)
#pragma unroll
      for (int cb = 0; cb < 4; ++cb)
        acc[rb][cb] = __builtin_amdgcn_wmma_f32_16x16x32_bf16(
            false, a[rb], false, b[cb], (short)0, acc[rb][cb], false, false);
  }

#pragma unroll
  for (int rb = 0; rb < 2; ++rb) {
#pragma unroll
    for (int cb = 0; cb < 4; ++cb) {
      const int n = (ncb0 + cb) * 16 + (lane & 15);
      const float bb = (MODE == 1) ? bias[n] : 0.0f;
#pragma unroll
      for (int r = 0; r < 8; ++r) {
        const int m = r + ((lane >> 4) << 3);            // C row within frag
        float v = acc[rb][cb][r] + bb;
        if (MODE == 1) v = tanhf(v);
        C[(size_t)(wrow0 + rb * 16 + m) * HID + n] = (bf16)v;
      }
    }
  }
}

// ---------------------------------------------------------------------------
// M-net head: L = h2m @ MW3 + Mb3 (N=9), Minv = L L^T + 0.1 I,
// p = Minv^{-1} qdot (Cramer). One wave per row (wave32 shfl reduction).
// ---------------------------------------------------------------------------
__global__ void k_mnet_head(const float* __restrict__ x,
                            const bf16* __restrict__ h2m,
                            const float* __restrict__ MW3,
                            const float* __restrict__ Mb3,
                            float* __restrict__ Lb, float* __restrict__ Mib,
                            float* __restrict__ pb)
{
  const int lane = threadIdx.x & 31;
  const int row  = blockIdx.x * 8 + (threadIdx.x >> 5);
  const bf16* h  = h2m + (size_t)row * HID;
  float L[9];
#pragma unroll
  for (int i = 0; i < 9; ++i) L[i] = 0.0f;
  for (int k = lane; k < HID; k += 32) {
    float hv = (float)h[k];
    const float* wv = MW3 + k * 9;
#pragma unroll
    for (int i = 0; i < 9; ++i) L[i] += hv * wv[i];
  }
#pragma unroll
  for (int s = 16; s > 0; s >>= 1) {
#pragma unroll
    for (int i = 0; i < 9; ++i) L[i] += __shfl_xor(L[i], s, 32);
  }
  if (lane == 0) {
#pragma unroll
    for (int i = 0; i < 9; ++i) L[i] += Mb3[i];
    float M[9];
#pragma unroll
    for (int r = 0; r < 3; ++r) {
#pragma unroll
      for (int cI = 0; cI < 3; ++cI) {
        float s = 0.0f;
#pragma unroll
        for (int a2 = 0; a2 < 3; ++a2) s += L[r * 3 + a2] * L[cI * 3 + a2];
        M[r * 3 + cI] = s + ((r == cI) ? 0.1f : 0.0f);
      }
    }
    float M00 = M[0], M01 = M[1], M02 = M[2];
    float M10 = M[3], M11 = M[4], M12 = M[5];
    float M20 = M[6], M21 = M[7], M22 = M[8];
    float adj00 = M11 * M22 - M12 * M21;
    float adj01 = M02 * M21 - M01 * M22;
    float adj02 = M01 * M12 - M02 * M11;
    float adj10 = M12 * M20 - M10 * M22;
    float adj11 = M00 * M22 - M02 * M20;
    float adj12 = M02 * M10 - M00 * M12;
    float adj20 = M10 * M21 - M11 * M20;
    float adj21 = M01 * M20 - M00 * M21;
    float adj22 = M00 * M11 - M01 * M10;
    float det = M00 * adj00 + M01 * adj10 + M02 * adj20;
    float id  = 1.0f / det;
    float q0 = x[(size_t)row * XDIM + 6];
    float q1 = x[(size_t)row * XDIM + 7];
    float q2 = x[(size_t)row * XDIM + 8];
    float p0 = (adj00 * q0 + adj01 * q1 + adj02 * q2) * id;
    float p1 = (adj10 * q0 + adj11 * q1 + adj12 * q2) * id;
    float p2 = (adj20 * q0 + adj21 * q1 + adj22 * q2) * id;
#pragma unroll
    for (int i = 0; i < 9; ++i) { Lb[(size_t)row * 9 + i] = L[i]; Mib[(size_t)row * 9 + i] = M[i]; }
    pb[(size_t)row * 3 + 0] = p0;
    pb[(size_t)row * 3 + 1] = p1;
    pb[(size_t)row * 3 + 2] = p2;
  }
}

// ---------------------------------------------------------------------------
// G-net head: g_q = h2g @ GW3 + Gb3 (N=3), F = g_q * u. One wave per row.
// ---------------------------------------------------------------------------
__global__ void k_gnet_head(const float* __restrict__ x,
                            const bf16* __restrict__ h2g,
                            const float* __restrict__ GW3,
                            const float* __restrict__ Gb3,
                            float* __restrict__ Fb)
{
  const int lane = threadIdx.x & 31;
  const int row  = blockIdx.x * 8 + (threadIdx.x >> 5);
  const bf16* h  = h2g + (size_t)row * HID;
  float s[3] = {0.0f, 0.0f, 0.0f};
  for (int k = lane; k < HID; k += 32) {
    float hv = (float)h[k];
#pragma unroll
    for (int i = 0; i < 3; ++i) s[i] += hv * GW3[k * 3 + i];
  }
#pragma unroll
  for (int sh = 16; sh > 0; sh >>= 1) {
#pragma unroll
    for (int i = 0; i < 3; ++i) s[i] += __shfl_xor(s[i], sh, 32);
  }
  if (lane == 0) {
    float u = x[(size_t)row * XDIM + 9];
#pragma unroll
    for (int i = 0; i < 3; ++i) Fb[(size_t)row * 3 + i] = (s[i] + Gb3[i]) * u;
  }
}

// ---------------------------------------------------------------------------
// M-backward seed: A_mb[row,k] = (1-h2m^2) * sum_i (p p^T L)_i * MW3[k,i]
// ---------------------------------------------------------------------------
__global__ void k_build_amb(const bf16* __restrict__ h2m,
                            const float* __restrict__ Lb,
                            const float* __restrict__ pb,
                            const float* __restrict__ MW3,
                            bf16* __restrict__ outA)
{
  int idx = blockIdx.x * blockDim.x + threadIdx.x;
  int k   = idx & (HID - 1);
  int row = idx >> 10;
  const float* L = Lb + (size_t)row * 9;
  const float* p = pb + (size_t)row * 3;
  float w[3];
#pragma unroll
  for (int j = 0; j < 3; ++j)
    w[j] = p[0] * L[0 * 3 + j] + p[1] * L[1 * 3 + j] + p[2] * L[2 * 3 + j];
  const float* m3 = MW3 + (size_t)k * 9;
  float s = 0.0f;
#pragma unroll
  for (int i = 0; i < 3; ++i) {
#pragma unroll
    for (int j = 0; j < 3; ++j) s += p[i] * w[j] * m3[i * 3 + j];
  }
  float hv = (float)h2m[idx];
  outA[idx] = (bf16)(s * (1.0f - hv * hv));
}

// V-backward seed: A_vb[row,k] = VW3[k] * (1 - h2v^2)
__global__ void k_build_avb(const bf16* __restrict__ h2v,
                            const float* __restrict__ VW3,
                            bf16* __restrict__ outA)
{
  int idx = blockIdx.x * blockDim.x + threadIdx.x;
  int k   = idx & (HID - 1);
  float hv = (float)h2v[idx];
  outA[idx] = (bf16)(VW3[k] * (1.0f - hv * hv));
}

// ---------------------------------------------------------------------------
// dH[row, c<6] (+)= sum_k gh1[row,k]*(1-h1^2)*W1[c,k]. One wave per row.
// ---------------------------------------------------------------------------
__global__ void k_dH(const bf16* __restrict__ gh1, const bf16* __restrict__ h1,
                     const float* __restrict__ W1, float* __restrict__ dH,
                     int accum)
{
  const int lane = threadIdx.x & 31;
  const int row  = blockIdx.x * 8 + (threadIdx.x >> 5);
  float s[6] = {0.0f, 0.0f, 0.0f, 0.0f, 0.0f, 0.0f};
  for (int k = lane; k < HID; k += 32) {
    float hv = (float)h1[(size_t)row * HID + k];
    float g  = (float)gh1[(size_t)row * HID + k] * (1.0f - hv * hv);
#pragma unroll
    for (int c = 0; c < 6; ++c) s[c] += g * W1[c * HID + k];
  }
#pragma unroll
  for (int sh = 16; sh > 0; sh >>= 1) {
#pragma unroll
    for (int c = 0; c < 6; ++c) s[c] += __shfl_xor(s[c], sh, 32);
  }
  if (lane == 0) {
#pragma unroll
    for (int c = 0; c < 6; ++c) {
      if (accum) dH[(size_t)row * 6 + c] += s[c];
      else       dH[(size_t)row * 6 + c]  = s[c];
    }
  }
}

// ---------------------------------------------------------------------------
// Finalize grads: dq = Minv p, dp = sin*dHcos - cos*dHsin + F,
// out[:,0:6] = (-sin*dq, cos*dq), out[:,9] = 0. Thread per row.
// ---------------------------------------------------------------------------
__global__ void k_finalize(const float* __restrict__ x,
                           const float* __restrict__ dH,
                           const float* __restrict__ Mib,
                           const float* __restrict__ pb,
                           const float* __restrict__ Fb,
                           float* __restrict__ out,
                           float* __restrict__ dpb,
                           float* __restrict__ tangb)
{
  int row = blockIdx.x * blockDim.x + threadIdx.x;
  float p[3], dq[3];
#pragma unroll
  for (int i = 0; i < 3; ++i) p[i] = pb[(size_t)row * 3 + i];
  const float* M = Mib + (size_t)row * 9;
#pragma unroll
  for (int i = 0; i < 3; ++i)
    dq[i] = M[i * 3 + 0] * p[0] + M[i * 3 + 1] * p[1] + M[i * 3 + 2] * p[2];
#pragma unroll
  for (int i = 0; i < 3; ++i) {
    float cq = x[(size_t)row * XDIM + i];
    float sq = x[(size_t)row * XDIM + 3 + i];
    float dpv = sq * dH[(size_t)row * 6 + i] - cq * dH[(size_t)row * 6 + 3 + i]
              + Fb[(size_t)row * 3 + i];
    dpb[(size_t)row * 3 + i] = dpv;
    float t0 = -sq * dq[i];
    float t1 =  cq * dq[i];
    out[(size_t)row * XDIM + i]     = t0;
    out[(size_t)row * XDIM + 3 + i] = t1;
    tangb[(size_t)row * 6 + i]     = t0;
    tangb[(size_t)row * 6 + 3 + i] = t1;
  }
  out[(size_t)row * XDIM + 9] = 0.0f;
}

// JVP layer-1: t1[row,j] = (1-h1m^2) * sum_c tang[c]*MW1[c,j]
__global__ void k_build_t1(const bf16* __restrict__ h1m,
                           const float* __restrict__ tangb,
                           const float* __restrict__ MW1,
                           bf16* __restrict__ outA)
{
  int idx = blockIdx.x * blockDim.x + threadIdx.x;
  int j   = idx & (HID - 1);
  int row = idx >> 10;
  float s = 0.0f;
#pragma unroll
  for (int c = 0; c < 6; ++c) s += tangb[(size_t)row * 6 + c] * MW1[c * HID + j];
  float hv = (float)h1m[idx];
  outA[idx] = (bf16)(s * (1.0f - hv * hv));
}

// ---------------------------------------------------------------------------
// JVP head: dL = ((1-h2m^2)*t2) @ MW3; dM = dL L^T + L dL^T;
// ddq = Minv dp + dM p; out[:,6:9] = ddq. One wave per row.
// ---------------------------------------------------------------------------
__global__ void k_jvp_head(const bf16* __restrict__ t2,
                           const bf16* __restrict__ h2m,
                           const float* __restrict__ MW3,
                           const float* __restrict__ Lb,
                           const float* __restrict__ Mib,
                           const float* __restrict__ pb,
                           const float* __restrict__ dpb,
                           float* __restrict__ out)
{
  const int lane = threadIdx.x & 31;
  const int row  = blockIdx.x * 8 + (threadIdx.x >> 5);
  float dL[9];
#pragma unroll
  for (int i = 0; i < 9; ++i) dL[i] = 0.0f;
  for (int k = lane; k < HID; k += 32) {
    float hv = (float)h2m[(size_t)row * HID + k];
    float dh = (float)t2[(size_t)row * HID + k] * (1.0f - hv * hv);
    const float* wv = MW3 + (size_t)k * 9;
#pragma unroll
    for (int i = 0; i < 9; ++i) dL[i] += dh * wv[i];
  }
#pragma unroll
  for (int sh = 16; sh > 0; sh >>= 1) {
#pragma unroll
    for (int i = 0; i < 9; ++i) dL[i] += __shfl_xor(dL[i], sh, 32);
  }
  if (lane == 0) {
    const float* L  = Lb  + (size_t)row * 9;
    const float* M  = Mib + (size_t)row * 9;
    const float* p  = pb  + (size_t)row * 3;
    const float* dp = dpb + (size_t)row * 3;
    float dM[9];
#pragma unroll
    for (int r = 0; r < 3; ++r) {
#pragma unroll
      for (int cI = 0; cI < 3; ++cI) {
        float s = 0.0f;
#pragma unroll
        for (int a2 = 0; a2 < 3; ++a2)
          s += dL[r * 3 + a2] * L[cI * 3 + a2] + L[r * 3 + a2] * dL[cI * 3 + a2];
        dM[r * 3 + cI] = s;
      }
    }
#pragma unroll
    for (int i = 0; i < 3; ++i) {
      float s = 0.0f;
#pragma unroll
      for (int j = 0; j < 3; ++j)
        s += M[i * 3 + j] * dp[j] + dM[i * 3 + j] * p[j];
      out[(size_t)row * XDIM + 6 + i] = s;
    }
  }
}

// ---------------------------------------------------------------------------
extern "C" void kernel_launch(void* const* d_in, const int* in_sizes, int n_in,
                              void* d_out, int out_size, void* d_ws, size_t ws_size,
                              hipStream_t stream)
{
  (void)in_sizes; (void)n_in; (void)out_size; (void)ws_size;
  const float* x   = (const float*)d_in[1];
  const float* MW1 = (const float*)d_in[2];
  const float* Mb1 = (const float*)d_in[3];
  const float* MW2 = (const float*)d_in[4];
  const float* Mb2 = (const float*)d_in[5];
  const float* MW3 = (const float*)d_in[6];
  const float* Mb3 = (const float*)d_in[7];
  const float* VW1 = (const float*)d_in[8];
  const float* Vb1 = (const float*)d_in[9];
  const float* VW2 = (const float*)d_in[10];
  const float* Vb2 = (const float*)d_in[11];
  const float* VW3 = (const float*)d_in[12];
  const float* GW1 = (const float*)d_in[14];
  const float* Gb1 = (const float*)d_in[15];
  const float* GW2 = (const float*)d_in[16];
  const float* Gb2 = (const float*)d_in[17];
  const float* GW3 = (const float*)d_in[18];
  const float* Gb3 = (const float*)d_in[19];
  float* out = (float*)d_out;

  char* w = (char*)d_ws;
  size_t off = 0;
  auto take = [&](size_t bytes) -> char* {
    off = (off + 255) & ~(size_t)255;
    char* p = w + off;
    off += bytes;
    return p;
  };

  bf16* MW2p  = (bf16*)take((size_t)HID * HID * 2);
  bf16* MW2Tp = (bf16*)take((size_t)HID * HID * 2);
  bf16* VW2p  = (bf16*)take((size_t)HID * HID * 2);
  bf16* VW2Tp = (bf16*)take((size_t)HID * HID * 2);
  bf16* GW2p  = (bf16*)take((size_t)HID * HID * 2);
  bf16* A0 = (bf16*)take((size_t)BS * HID * 2);   // h1m (live to end)
  bf16* A1 = (bf16*)take((size_t)BS * HID * 2);   // h1v
  bf16* A2 = (bf16*)take((size_t)BS * HID * 2);   // h1g -> amb -> avb -> t1
  bf16* A3 = (bf16*)take((size_t)BS * HID * 2);   // h2m (live to end)
  bf16* A4 = (bf16*)take((size_t)BS * HID * 2);   // h2v
  bf16* A5 = (bf16*)take((size_t)BS * HID * 2);   // h2g -> gh1m -> gh1v -> t2
  float* Lb    = (float*)take((size_t)BS * 9 * 4);
  float* Mib   = (float*)take((size_t)BS * 9 * 4);
  float* pb    = (float*)take((size_t)BS * 3 * 4);
  float* Fb    = (float*)take((size_t)BS * 3 * 4);
  float* dHb   = (float*)take((size_t)BS * 6 * 4);
  float* dpb   = (float*)take((size_t)BS * 3 * 4);
  float* tangb = (float*)take((size_t)BS * 6 * 4);

  const int PACK_G = HID * HID / 256;
  k_pack<<<PACK_G, 256, 0, stream>>>(MW2, MW2p, 0);
  k_pack<<<PACK_G, 256, 0, stream>>>(MW2, MW2Tp, 1);
  k_pack<<<PACK_G, 256, 0, stream>>>(VW2, VW2p, 0);
  k_pack<<<PACK_G, 256, 0, stream>>>(VW2, VW2Tp, 1);
  k_pack<<<PACK_G, 256, 0, stream>>>(GW2, GW2p, 0);

  const int EW_G = BS * HID / 256;   // elementwise grids over (row, j)
  k_layer1<<<EW_G, 256, 0, stream>>>(x, MW1, Mb1, VW1, Vb1, GW1, Gb1, A0, A1, A2);

  dim3 gg(BS / 128, HID / 128);      // 256 x 8 workgroups of 8 waves
  k_gemm<1><<<gg, 256, 0, stream>>>(A0, MW2p, Mb2, A3);   // h2m
  k_gemm<1><<<gg, 256, 0, stream>>>(A1, VW2p, Vb2, A4);   // h2v
  k_gemm<1><<<gg, 256, 0, stream>>>(A2, GW2p, Gb2, A5);   // h2g

  k_mnet_head<<<BS / 8, 256, 0, stream>>>(x, A3, MW3, Mb3, Lb, Mib, pb);
  k_gnet_head<<<BS / 8, 256, 0, stream>>>(x, A5, GW3, Gb3, Fb);

  // M-net backward through W2^T
  k_build_amb<<<EW_G, 256, 0, stream>>>(A3, Lb, pb, MW3, A2);
  k_gemm<0><<<gg, 256, 0, stream>>>(A2, MW2Tp, nullptr, A5);   // gh1m
  k_dH<<<BS / 8, 256, 0, stream>>>(A5, A0, MW1, dHb, 0);

  // V-net backward through W2^T
  k_build_avb<<<EW_G, 256, 0, stream>>>(A4, VW3, A2);
  k_gemm<0><<<gg, 256, 0, stream>>>(A2, VW2Tp, nullptr, A5);   // gh1v
  k_dH<<<BS / 8, 256, 0, stream>>>(A5, A1, VW1, dHb, 1);

  k_finalize<<<BS / 256, 256, 0, stream>>>(x, dHb, Mib, pb, Fb, out, dpb, tangb);

  // JVP through M-net
  k_build_t1<<<EW_G, 256, 0, stream>>>(A0, tangb, MW1, A2);
  k_gemm<0><<<gg, 256, 0, stream>>>(A2, MW2p, nullptr, A5);    // t2
  k_jvp_head<<<BS / 8, 256, 0, stream>>>(A5, A3, MW3, Lb, Mib, pb, dpb, out);
}
